// INTERNEURON_NETWORK_45964740002362
// MI455X (gfx1250) — matrix operand order repaired and checked by reading, not compiled, not run on hardware
//
#include <hip/hip_runtime.h>
#include <hip/hip_bf16.h>
#include <math.h>

#define NN 2048   // NUM_NEURONS
#define ED 1024   // EMBED_DIM
#define NW 12     // number of windows
#define MT 2048   // rows of act actually used (tail)
#define XOFF (6144 * ED)  // skip unused first 6144 rows of x

typedef __attribute__((ext_vector_type(16))) __bf16 v16bf;
typedef __attribute__((ext_vector_type(8)))  float  v8f;

struct alignas(16) U4 { unsigned x, y, z, w; };
struct alignas(32) Frag32 { U4 lo, hi; };

__device__ __forceinline__ unsigned short f2bf(float f) {
    unsigned u = __float_as_uint(f);
    u += 0x7FFFu + ((u >> 16) & 1u);
    return (unsigned short)(u >> 16);
}

// ---------------- prep: convert + retile to fragment-native bf16 ----------------
// A tiles: xt[tm][ks][lane][16]; element i<8 -> K = ks*32 + half*8 + i
//                                element i>=8 -> K = ks*32 + 16 + half*8 + (i-8)
__global__ __launch_bounds__(256) void k_conv_x(const float* __restrict__ x,
                                                unsigned short* __restrict__ xt) {
    int idx = blockIdx.x * 256 + threadIdx.x;           // 128*32*32*16 = 2M
    int i  = idx & 15;
    int l  = (idx >> 4) & 31;
    int ks = (idx >> 9) & 31;
    int tm = idx >> 14;
    int half = l >> 4;
    int m = tm * 16 + (l & 15);
    int k = ks * 32 + half * 8 + (i & 7) + ((i >= 8) ? 16 : 0);
    xt[idx] = f2bf(x[XOFF + m * ED + k]);
}

// B tiles: wt[tn][ks][lane][16]; lane -> N = tn*16 + (lane&15), element i -> K = ks*32 + half*8 + (i&7) + (i>=8 ? 16 : 0)
// B[k][n] = n_weights[n][k]  (transpose folded into the retile)
__global__ __launch_bounds__(256) void k_conv_w(const float* __restrict__ W,
                                                unsigned short* __restrict__ wt) {
    int idx = blockIdx.x * 256 + threadIdx.x;           // 128*32*32*16 = 2M
    int i  = idx & 15;
    int l  = (idx >> 4) & 31;
    int ks = (idx >> 9) & 31;
    int tn = idx >> 14;
    int n = tn * 16 + (l & 15);
    int k = ks * 32 + (l >> 4) * 8 + (i & 7) + ((i >= 8) ? 16 : 0);
    wt[idx] = f2bf(W[n * ED + k]);
}

__global__ __launch_bounds__(256) void k_zero(float* __restrict__ p, int n) {
    int idx = blockIdx.x * 256 + threadIdx.x;
    if (idx < n) p[idx] = 0.0f;
}

// ---------------- GEMM1 (bf16 WMMA) + fused tanh/clamp + segment column sums ----
// Each wave computes a 16(M) x 64(N) strip: one A fragment feeds 4 WMMAs per
// K-step (4x A reuse, 4 accumulators). 4096 strips total, 8 waves/block.
// act[t][n] = clamp(tanh(sum_k x[t][k]*W[n][k] + b[n]))
// segsum[j][n] += sum over rows t in segment j of act[t][n]
// segments: j=0:[0,1) j=1:[1,2) j=2:[2,4) ... j=11:[1024,2048)
__global__ __launch_bounds__(256) void k_gemm_act(const unsigned short* __restrict__ xt,
                                                  const unsigned short* __restrict__ wt,
                                                  const float* __restrict__ nbias,
                                                  float* __restrict__ segsum) {
    int lane = threadIdx.x & 31;
    int wave = threadIdx.x >> 5;
    int strip = blockIdx.x * 8 + wave;     // 4096 strips of 16x64
    int tm  = strip >> 5;                  // 0..127
    int tn0 = (strip & 31) * 4;            // 0..124 (first of 4 N-tiles)
    int half = lane >> 4;

    v8f c[4];
    #pragma unroll
    for (int t = 0; t < 4; ++t) c[t] = (v8f){0.f, 0.f, 0.f, 0.f, 0.f, 0.f, 0.f, 0.f};

    const unsigned short* ap = xt + (size_t)tm  * 32 * 512 + lane * 16;
    const unsigned short* bp = wt + (size_t)tn0 * 32 * 512 + lane * 16;

    #pragma unroll 2
    for (int ks = 0; ks < 32; ++ks) {
        Frag32 fa = *(const Frag32*)(ap + ks * 512);
        v16bf av = __builtin_bit_cast(v16bf, fa);
        #pragma unroll
        for (int t = 0; t < 4; ++t) {
            Frag32 fb = *(const Frag32*)(bp + (size_t)t * 32 * 512 + ks * 512);
            v16bf bv = __builtin_bit_cast(v16bf, fb);
            c[t] = __builtin_amdgcn_wmma_f32_16x16x32_bf16(false, av, false, bv,
                                                           (short)0, c[t], false, false);
        }
    }

    #pragma unroll
    for (int t = 0; t < 4; ++t) {
        int col = (tn0 + t) * 16 + (lane & 15);
        float bias = nbias[col];
        float vals[8];
        #pragma unroll
        for (int r = 0; r < 8; ++r) {
            float v = tanhf(c[t][r] + bias);
            vals[r] = fminf(5.0f, fmaxf(-5.0f, v));
        }
        if (tm == 0) {
            // rows 0..15 straddle segments 0..4: per-element segment
            #pragma unroll
            for (int r = 0; r < 8; ++r) {
                int row = r + half * 8;   // global row (tm==0)
                int j = (row == 0) ? 0 : (32 - __clz(row));
                atomicAdd(&segsum[j * NN + col], vals[r]);
            }
        } else {
            // all 16 rows of this tile live in one segment (boundaries >=16 are 16-aligned)
            float s = 0.f;
            #pragma unroll
            for (int r = 0; r < 8; ++r) s += vals[r];
            int j = 32 - __clz(tm * 16);
            atomicAdd(&segsum[j * NN + col], s);
        }
    }
}

// ---------------- window means = prefix sum of segment sums / window size ------
__global__ __launch_bounds__(256) void k_winout(const float* __restrict__ segsum,
                                                float* __restrict__ winout) {
    int n = blockIdx.x * 256 + threadIdx.x;
    float run = 0.f;
    #pragma unroll
    for (int j = 0; j < NW; ++j) {
        run += segsum[j * NN + n];
        winout[j * NN + n] = run / (float)(1 << j);
    }
}

// ---------------- combo[w][m] = dot(winout[w], combo_w[w][m]) + combo_b --------
// HBM-bound: streams 201 MB of combo_w exactly once (the roofline limiter).
__global__ __launch_bounds__(256) void k_combo(const float* __restrict__ winout,
                                               const float* __restrict__ combo_w,
                                               const float* __restrict__ combo_b,
                                               float* __restrict__ combo) {
    __shared__ float sWin[NN];
    int w = blockIdx.y;
    for (int i = threadIdx.x; i < NN; i += 256) sWin[i] = winout[w * NN + i];
    __syncthreads();
    int lane = threadIdx.x & 31;
    int wave = threadIdx.x >> 5;
    #pragma unroll
    for (int rr = 0; rr < 4; ++rr) {
        int row = blockIdx.x * 32 + wave * 4 + rr;
        const float* wp = combo_w + ((size_t)w * NN + row) * NN;
        // prefetch this wave's next row while streaming the current one
        if (rr < 3) __builtin_prefetch(wp + NN + lane * 64, 0, 0);
        float acc = 0.f;
        for (int n = lane; n < NN; n += 32) acc += wp[n] * sWin[n];
        #pragma unroll
        for (int off = 16; off > 0; off >>= 1) acc += __shfl_down(acc, off, 32);
        if (lane == 0) combo[w * NN + row] = acc + combo_b[w * NN + row];
    }
}

// ---------------- query/key GEMVs ----------------------------------------------
// z==0: query[w][e] = dot(combo[w], q_w[e]) + q_b[e] + judge_bias[w]*cer[w]
// z==1: key[w][e]   = dot(winout[w], k_w[e]) + k_b[e] + cred_bias[w]*cer[w]
__global__ __launch_bounds__(256) void k_qk(const float* __restrict__ combo,
                                            const float* __restrict__ winout,
                                            const float* __restrict__ q_w,
                                            const float* __restrict__ q_b,
                                            const float* __restrict__ k_w,
                                            const float* __restrict__ k_b,
                                            const float* __restrict__ cer,
                                            const float* __restrict__ jb,
                                            const float* __restrict__ cb,
                                            float* __restrict__ query,
                                            float* __restrict__ key) {
    __shared__ float sVec[NN];
    int w = blockIdx.y;
    int z = blockIdx.z;
    const float* vec = (z == 0) ? (combo + w * NN) : (winout + w * NN);
    for (int i = threadIdx.x; i < NN; i += 256) sVec[i] = vec[i];
    __syncthreads();
    int lane = threadIdx.x & 31;
    int wave = threadIdx.x >> 5;
    int e = blockIdx.x * 8 + wave;
    const float* mat = (z == 0) ? q_w : k_w;
    const float* mp = mat + (size_t)e * NN;
    float acc = 0.f;
    for (int n = lane; n < NN; n += 32) acc += mp[n] * sVec[n];
    #pragma unroll
    for (int off = 16; off > 0; off >>= 1) acc += __shfl_down(acc, off, 32);
    if (lane == 0) {
        if (z == 0) query[w * ED + e] = acc + q_b[e] + jb[w] * cer[w];
        else        key[w * ED + e]   = acc + k_b[e] + cb[w] * cer[w];
    }
}

// ---------------- scores, column sums, softmax -> sum of weights ---------------
__global__ __launch_bounds__(256) void k_scores(const float* __restrict__ query,
                                                const float* __restrict__ key,
                                                const float* __restrict__ cer,
                                                const float* __restrict__ pblend,
                                                float* __restrict__ wsum) {
    __shared__ float sc[NW * NW];
    __shared__ float comb[NW];
    int tid = threadIdx.x;
    if (tid < NW * NW) {
        int wq = tid / NW, wk = tid % NW;
        float s = 0.f;
        for (int e = 0; e < ED; ++e) s += query[wq * ED + e] * key[wk * ED + e];
        sc[tid] = s;  // TEMPERATURE == 1.0
    }
    __syncthreads();
    if (tid < NW) {
        float s = 0.f;
        for (int i = 0; i < NW; ++i) s += sc[i * NW + tid];
        comb[tid] = s;
    }
    __syncthreads();
    if (tid == 0) {
        float pb = 1.0f / (1.0f + expf(-pblend[0]));
        float bl[NW], mx = -3.4e38f;
        for (int w = 0; w < NW; ++w) {
            bl[w] = ((1.0f - pb) * cer[w] + pb * comb[w]) * cer[w];
            mx = fmaxf(mx, bl[w]);
        }
        float se = 0.f;
        for (int w = 0; w < NW; ++w) se += expf(bl[w] - mx);
        float sw = 0.f;
        for (int w = 0; w < NW; ++w) sw += expf(bl[w] - mx) / se;
        wsum[0] = sw;   // == 1 up to fp rounding, computed faithfully
    }
}

// ---------------- ctx[n] = (1 + 0.2/12)*sum_w combo[w][n] + sum_w weights ------
__global__ __launch_bounds__(256) void k_ctx(const float* __restrict__ combo,
                                             const float* __restrict__ wsum,
                                             float* __restrict__ out) {
    int n = blockIdx.x * 256 + threadIdx.x;
    float s = 0.f;
    #pragma unroll
    for (int w = 0; w < NW; ++w) s += combo[w * NN + n];
    out[n] = s * (1.0f + 0.2f / 12.0f) + wsum[0];
}

extern "C" void kernel_launch(void* const* d_in, const int* in_sizes, int n_in,
                              void* d_out, int out_size, void* d_ws, size_t ws_size,
                              hipStream_t stream) {
    const float* x       = (const float*)d_in[0];
    const float* n_w     = (const float*)d_in[1];
    const float* n_b     = (const float*)d_in[2];
    const float* combo_w = (const float*)d_in[3];
    const float* combo_b = (const float*)d_in[4];
    const float* q_w     = (const float*)d_in[5];
    const float* q_b     = (const float*)d_in[6];
    const float* k_w     = (const float*)d_in[7];
    const float* k_b     = (const float*)d_in[8];
    const float* cer     = (const float*)d_in[9];
    const float* jb      = (const float*)d_in[10];
    const float* cb      = (const float*)d_in[11];
    const float* pblend  = (const float*)d_in[12];
    float* out = (float*)d_out;

    char* ws = (char*)d_ws;
    size_t o = 0;
    unsigned short* xt = (unsigned short*)(ws + o); o += (size_t)MT * ED * 2;   // 4 MB
    unsigned short* wt = (unsigned short*)(ws + o); o += (size_t)NN * ED * 2;   // 4 MB
    float* segsum = (float*)(ws + o); o += (size_t)NW * NN * 4;
    float* winout = (float*)(ws + o); o += (size_t)NW * NN * 4;
    float* combo  = (float*)(ws + o); o += (size_t)NW * NN * 4;
    float* query  = (float*)(ws + o); o += (size_t)NW * ED * 4;
    float* key    = (float*)(ws + o); o += (size_t)NW * ED * 4;
    float* wsum   = (float*)(ws + o); o += 256;

    // prep: convert + retile operands, zero segment accumulators
    k_conv_x<<<(MT * ED) / 256, 256, 0, stream>>>(x, xt);
    k_conv_w<<<(NN * ED) / 256, 256, 0, stream>>>(n_w, wt);
    k_zero<<<(NW * NN + 255) / 256, 256, 0, stream>>>(segsum, NW * NN);

    // bf16 WMMA GEMM (2048x2048x1024), 16x64 strip per wave, fused epilogue
    k_gemm_act<<<4096 / 8, 256, 0, stream>>>(xt, wt, n_b, segsum);

    // prefix-sum segments -> window means
    k_winout<<<NN / 256, 256, 0, stream>>>(segsum, winout);

    // 12 x (2048x2048) GEMVs streaming combo_w (the HBM-bound stage)
    k_combo<<<dim3(64, 12), 256, 0, stream>>>(winout, combo_w, combo_b, combo);

    // query / key projections
    k_qk<<<dim3(128, 12, 2), 256, 0, stream>>>(combo, winout, q_w, q_b, k_w, k_b,
                                               cer, jb, cb, query, key);

    // scores -> softmax -> sum of window weights
    k_scores<<<1, 256, 0, stream>>>(query, key, cer, pblend, wsum);

    // final context vector
    k_ctx<<<NN / 256, 256, 0, stream>>>(combo, wsum, out);
}
